// FocusedGQAttention_9259949490450
// MI455X (gfx1250) — matrix operand order repaired and checked by reading, not compiled, not run on hardware
//
#include <hip/hip_runtime.h>
#include <hip/hip_bf16.h>

// Focused linear GQ attention for MI455X (gfx1250, wave32, WMMA).
// Big GEMMs on v_wmma_f32_16x16x32_f16; tile staging via async global->LDS
// (ASYNCcnt) with double buffering so DMA overlaps the WMMA bursts.

typedef __attribute__((ext_vector_type(16))) _Float16 v16h;
typedef __attribute__((ext_vector_type(8)))  float    v8f;

#define TOKENS 32768   // B*N
#define DIMC   1152
#define NQKV   1920    // 1152 q + 768 kv
#define NHEAD  12
#define NKV    4
#define HEADD  96
#define SEQ    1024
#define NBATCH 32

// ---------------------------------------------------------------------------
// Async global->LDS copy (GLOBAL_LOAD_ASYNC_TO_LDS_B128, ASYNCcnt) with a
// synchronous fallback if the builtin is unavailable on this toolchain.
// Builtin signature (from compiler diagnostic): (int4 AS1*, int4 AS3*, Ii, Ii).
// ---------------------------------------------------------------------------
#if defined(__has_builtin)
#if __has_builtin(__builtin_amdgcn_global_load_async_to_lds_b128)
#define HAVE_ASYNC_LDS 1
#endif
#if __has_builtin(__builtin_amdgcn_s_wait_asynccnt)
#define HAVE_WAIT_ASYNC 1
#endif
#endif

typedef __attribute__((vector_size(16))) int i4v;
typedef __attribute__((address_space(1))) i4v* gptr_b128;
typedef __attribute__((address_space(3))) i4v* lptr_b128;

static __device__ __forceinline__ void async_copy16(const _Float16* __restrict__ g,
                                                    _Float16* __restrict__ l) {
#if defined(HAVE_ASYNC_LDS)
  __builtin_amdgcn_global_load_async_to_lds_b128(
      (gptr_b128)(i4v*)const_cast<_Float16*>(g),
      (lptr_b128)(i4v*)l, 0, 0);
#else
  *reinterpret_cast<float4*>(l) = *reinterpret_cast<const float4*>(g);
#endif
}

static __device__ __forceinline__ void wait_async_all() {
#if defined(HAVE_ASYNC_LDS)
#if defined(HAVE_WAIT_ASYNC)
  __builtin_amdgcn_s_wait_asynccnt(0);
#else
  asm volatile("s_wait_asynccnt 0" ::: "memory");
#endif
#endif
}

// ---------------------------------------------------------------------------
// WMMA fragment loaders (ISA 7.12.2, wave32).
// A 16x32 f16: lane L (r=L&15, h=L>>4) holds row M=r,
//   halves 0..7  -> K = h*8 + 0..7 ; halves 8..15 -> K = 16 + h*8 + 0..7
// ---------------------------------------------------------------------------
static __device__ __forceinline__ v16h frag_load_a(const _Float16* __restrict__ base,
                                                   int ld, int lane) {
  const int r = lane & 15;
  const int h = lane >> 4;
  const _Float16* p = base + (size_t)r * ld + h * 8;
  v16h out;
#pragma unroll
  for (int i = 0; i < 8; ++i) { out[i] = p[i]; out[8 + i] = p[16 + i]; }
  return out;
}

// B 32x16 f16 staged as [n][k] in LDS: lane L (n=L&15, h=L>>4) holds
// column n, halves 0..7 -> K = h*8 + 0..7 ; halves 8..15 -> K = 16 + h*8 + 0..7
static __device__ __forceinline__ v16h frag_load_b(const _Float16* __restrict__ base,
                                                   int ld, int lane) {
  const int n = lane & 15;
  const int h = lane >> 4;
  const _Float16* p = base + (size_t)n * ld + h * 8;
  v16h out;
#pragma unroll
  for (int i = 0; i < 8; ++i) { out[i] = p[i]; out[8 + i] = p[16 + i]; }
  return out;
}

// ---------------------------------------------------------------------------
// Large GEMM: C[M][N] = A[M][K](f16) * Bt[N][K](f16, B transposed) + bias[N]
// 128x128 block tile, BK=32, 256 threads = 8 waves (4 along M x 2 along N),
// each wave owns a 32x64 patch = 2x4 WMMA tiles. Double-buffered async
// staging: tile k+1 DMA overlaps the 8-WMMA burst on tile k.
// ---------------------------------------------------------------------------
__global__ __launch_bounds__(256) void gemm_f16_f32(
    const _Float16* __restrict__ A, const _Float16* __restrict__ Bt,
    const float* __restrict__ bias, float* __restrict__ C,
    int M, int Ncols, int K) {
  __shared__ _Float16 sA[2][128 * 32];
  __shared__ _Float16 sB[2][128 * 32];
  const int tid  = threadIdx.x;
  const int wave = tid >> 5, lane = tid & 31;
  const int wm = wave >> 1, wn = wave & 1;
  const int blkM = blockIdx.y * 128, blkN = blockIdx.x * 128;

  auto stage = [&](int buf, int k0) {
#pragma unroll
    for (int i = tid; i < 512; i += 256) {   // 128 rows x 32 halves, 16B chunks
      const int r = i >> 2, c8 = (i & 3) * 8;
      async_copy16(&A[(size_t)(blkM + r) * K + k0 + c8], &sA[buf][r * 32 + c8]);
      async_copy16(&Bt[(size_t)(blkN + r) * K + k0 + c8], &sB[buf][r * 32 + c8]);
    }
  };

  const v8f vzero = {0.f, 0.f, 0.f, 0.f, 0.f, 0.f, 0.f, 0.f};
  v8f acc[2][4];
#pragma unroll
  for (int i = 0; i < 2; ++i)
#pragma unroll
    for (int j = 0; j < 4; ++j) acc[i][j] = vzero;

  stage(0, 0);
  wait_async_all();
  __syncthreads();

  int buf = 0;
  for (int k0 = 0; k0 < K; k0 += 32) {
    if (k0 + 32 < K) stage(buf ^ 1, k0 + 32);      // async prefetch next tile

    v16h af[2], bf[4];
#pragma unroll
    for (int i = 0; i < 2; ++i)
      af[i] = frag_load_a(&sA[buf][(wm * 32 + i * 16) * 32], 32, lane);
#pragma unroll
    for (int j = 0; j < 4; ++j)
      bf[j] = frag_load_b(&sB[buf][(wn * 64 + j * 16) * 32], 32, lane);
#pragma unroll
    for (int i = 0; i < 2; ++i)
#pragma unroll
      for (int j = 0; j < 4; ++j)
        acc[i][j] = __builtin_amdgcn_wmma_f32_16x16x32_f16(
            false, af[i], false, bf[j], (short)0, acc[i][j], false, false);

    wait_async_all();      // next tile landed in LDS
    __syncthreads();       // and everyone is done reading this tile
    buf ^= 1;
  }

  const int nn = lane & 15, hh = lane >> 4;   // C/D: n=lane%16, M split by lane half
#pragma unroll
  for (int i = 0; i < 2; ++i)
#pragma unroll
    for (int j = 0; j < 4; ++j) {
      const int col = blkN + wn * 64 + j * 16 + nn;
      const float bv = bias ? bias[col] : 0.0f;
#pragma unroll
      for (int r = 0; r < 8; ++r) {
        const int row = blkM + wm * 32 + i * 16 + hh * 8 + r;
        C[(size_t)row * Ncols + col] = acc[i][j][r] + bv;
      }
    }
}

// ---------------------------------------------------------------------------
// kv = k^T v per (b, kv-head): [96x1024] * [1024x96] -> stored transposed
// [e][d] f16 so it can be a B-operand for the attn GEMM. 6 waves, BK=64,
// double-buffered async staging.
// ---------------------------------------------------------------------------
__global__ __launch_bounds__(192) void gemm_kv_kernel(
    const _Float16* __restrict__ KT, const _Float16* __restrict__ VT,
    _Float16* __restrict__ kvT) {
  __shared__ _Float16 sK[2][96 * 64];
  __shared__ _Float16 sV[2][96 * 64];
  const int bk = blockIdx.x;                    // b*4 + kvh
  const int tid = threadIdx.x;
  const int wave = tid >> 5, lane = tid & 31;
  const _Float16* kbase = KT + (size_t)bk * HEADD * SEQ;
  const _Float16* vbase = VT + (size_t)bk * HEADD * SEQ;

  auto stage = [&](int buf, int k0) {
    for (int i = tid; i < 768; i += 192) {      // 96 rows x 64 halves, 16B chunks
      const int r = i >> 3, c8 = (i & 7) * 8;
      async_copy16(&kbase[(size_t)r * SEQ + k0 + c8], &sK[buf][r * 64 + c8]);
      async_copy16(&vbase[(size_t)r * SEQ + k0 + c8], &sV[buf][r * 64 + c8]);
    }
  };

  const v8f vzero = {0.f, 0.f, 0.f, 0.f, 0.f, 0.f, 0.f, 0.f};
  v8f acc[6];
#pragma unroll
  for (int j = 0; j < 6; ++j) acc[j] = vzero;

  stage(0, 0);
  wait_async_all();
  __syncthreads();

  int buf = 0;
  for (int k0 = 0; k0 < SEQ; k0 += 64) {
    if (k0 + 64 < SEQ) stage(buf ^ 1, k0 + 64);
#pragma unroll
    for (int kk = 0; kk < 64; kk += 32) {
      const v16h a = frag_load_a(&sK[buf][(wave * 16) * 64 + kk], 64, lane);
#pragma unroll
      for (int j = 0; j < 6; ++j) {
        const v16h b = frag_load_b(&sV[buf][(j * 16) * 64 + kk], 64, lane);
        acc[j] = __builtin_amdgcn_wmma_f32_16x16x32_f16(
            false, a, false, b, (short)0, acc[j], false, false);
      }
    }
    wait_async_all();
    __syncthreads();
    buf ^= 1;
  }

  const int nn = lane & 15, hh = lane >> 4;
  _Float16* obase = kvT + (size_t)bk * HEADD * HEADD;
#pragma unroll
  for (int j = 0; j < 6; ++j) {
    const int e = j * 16 + nn;
#pragma unroll
    for (int r = 0; r < 8; ++r) {
      const int d = wave * 16 + hh * 8 + r;
      obase[(size_t)e * HEADD + d] = (_Float16)acc[j][r];   // transposed store
    }
  }
}

// ---------------------------------------------------------------------------
// attn = q @ kv, + depthwise-conv v, repacked f16 for the output projection.
// grid = (SEQ/128, B*H); whole B operand (96x96) + 128x96 A tile in LDS,
// staged once via async copies.
// ---------------------------------------------------------------------------
__global__ __launch_bounds__(256) void gemm_attn_kernel(
    const _Float16* __restrict__ Qh, const _Float16* __restrict__ kvT,
    const float* __restrict__ vdwc, _Float16* __restrict__ Oh) {
  __shared__ _Float16 sQ[128 * 96];
  __shared__ _Float16 sKV[96 * 96];
  const int bh = blockIdx.y;                    // b*12 + h
  const int b = bh / NHEAD, h = bh % NHEAD, kvh = h & 3;
  const int tid = threadIdx.x;
  const int wave = tid >> 5, lane = tid & 31;
  const int row0 = blockIdx.x * 128;

  const _Float16* qbase  = Qh + ((size_t)bh * SEQ + row0) * HEADD;     // contiguous
  const _Float16* kvbase = kvT + (size_t)(b * NKV + kvh) * HEADD * HEADD;
  for (int i = tid; i < 1536; i += 256)
    async_copy16(&qbase[(size_t)i * 8], &sQ[i * 8]);
  for (int i = tid; i < 1152; i += 256)
    async_copy16(&kvbase[(size_t)i * 8], &sKV[i * 8]);
  wait_async_all();
  __syncthreads();

  const v8f vzero = {0.f, 0.f, 0.f, 0.f, 0.f, 0.f, 0.f, 0.f};
  v8f acc[6];
#pragma unroll
  for (int j = 0; j < 6; ++j) acc[j] = vzero;
#pragma unroll
  for (int k0 = 0; k0 < 96; k0 += 32) {
    const v16h a = frag_load_a(&sQ[(wave * 16) * 96 + k0], 96, lane);
#pragma unroll
    for (int j = 0; j < 6; ++j) {
      const v16h bfr = frag_load_b(&sKV[(j * 16) * 96 + k0], 96, lane);
      acc[j] = __builtin_amdgcn_wmma_f32_16x16x32_f16(
          false, a, false, bfr, (short)0, acc[j], false, false);
    }
  }

  const int nn = lane & 15, hh = lane >> 4;
#pragma unroll
  for (int j = 0; j < 6; ++j) {
    const int e = j * 16 + nn;
#pragma unroll
    for (int r = 0; r < 8; ++r) {
      const int tok = row0 + wave * 16 + hh * 8 + r;
      const float v = acc[j][r] +
          vdwc[((size_t)(b * NKV + kvh) * SEQ + tok) * HEADD + e];
      Oh[((size_t)b * SEQ + tok) * DIMC + h * HEADD + e] = (_Float16)v;
    }
  }
}

// ---------------------------------------------------------------------------
// Elementwise / reduction / conv helper kernels
// ---------------------------------------------------------------------------
__global__ void f32_to_f16_vec(const float* __restrict__ src,
                               _Float16* __restrict__ dst, int n4) {
  const int stride = gridDim.x * blockDim.x;
  for (int i = blockIdx.x * blockDim.x + threadIdx.x; i < n4; i += stride) {
    const float4 v = reinterpret_cast<const float4*>(src)[i];
    dst[(size_t)i * 4 + 0] = (_Float16)v.x;
    dst[(size_t)i * 4 + 1] = (_Float16)v.y;
    dst[(size_t)i * 4 + 2] = (_Float16)v.z;
    dst[(size_t)i * 4 + 3] = (_Float16)v.w;
  }
}

// src[rows][cols] (f32) -> dst[cols][rows] (f16)
__global__ void transpose_to_f16(const float* __restrict__ src,
                                 _Float16* __restrict__ dst, int rows, int cols) {
  const int total = rows * cols;
  const int stride = gridDim.x * blockDim.x;
  for (int i = blockIdx.x * blockDim.x + threadIdx.x; i < total; i += stride) {
    const int o = i / rows, r = i % rows;
    dst[i] = (_Float16)src[(size_t)r * cols + o];
  }
}

__global__ void pack_bias_kernel(const float* __restrict__ wqb,
                                 const float* __restrict__ wkvb,
                                 float* __restrict__ biasC) {
  const int i = blockIdx.x * blockDim.x + threadIdx.x;
  if (i < NQKV) biasC[i] = (i < DIMC) ? wqb[i] : wkvb[i - DIMC];
}

// One block per (b,head): scale = sqrt(sum relu^2 / sum relu^6).
// blocks 0..383 = q heads, 384..511 = k heads.
__global__ __launch_bounds__(256) void reduce_scales_kernel(
    const float* __restrict__ qkv, float* __restrict__ scales) {
  const int blk = blockIdx.x;
  int b, colbase;
  if (blk < NBATCH * NHEAD) { b = blk / NHEAD; colbase = (blk % NHEAD) * HEADD; }
  else { const int t = blk - NBATCH * NHEAD; b = t >> 2; colbase = DIMC + (t & 3) * HEADD; }
  float s2 = 0.f, s6 = 0.f;
  for (int i = threadIdx.x; i < SEQ * HEADD; i += 256) {
    const int n = i / HEADD, d = i % HEADD;
    float v = qkv[((size_t)(b * SEQ + n)) * NQKV + colbase + d];
    v = fmaxf(v, 0.f);
    const float v2 = v * v, v3 = v2 * v;
    s2 += v2; s6 += v3 * v3;
  }
  __shared__ float r2[256], r6[256];
  r2[threadIdx.x] = s2; r6[threadIdx.x] = s6;
  __syncthreads();
  for (int off = 128; off > 0; off >>= 1) {
    if (threadIdx.x < off) {
      r2[threadIdx.x] += r2[threadIdx.x + off];
      r6[threadIdx.x] += r6[threadIdx.x + off];
    }
    __syncthreads();
  }
  if (threadIdx.x == 0) scales[blk] = (r6[0] > 0.f) ? sqrtf(r2[0] / r6[0]) : 0.f;
}

// q_focused -> Qh[bh][n][d] f16
__global__ void focus_q_kernel(const float* __restrict__ qkv,
                               const float* __restrict__ scales,
                               _Float16* __restrict__ Qh) {
  const int total = NBATCH * NHEAD * SEQ * HEADD;
  const int stride = gridDim.x * blockDim.x;
  for (int i = blockIdx.x * blockDim.x + threadIdx.x; i < total; i += stride) {
    const int bh = i / (SEQ * HEADD);
    const int rem = i % (SEQ * HEADD);
    const int n = rem / HEADD, d = rem % HEADD;
    const int b = bh / NHEAD, h = bh % NHEAD;
    float v = qkv[((size_t)(b * SEQ + n)) * NQKV + h * HEADD + d];
    v = fmaxf(v, 0.f);
    Qh[i] = (_Float16)(scales[bh] * v * v * v);
  }
}

// k_focused -> KT[bk][d][n], v -> VT[bk][e][n]  (both transposed, f16)
__global__ void pack_kv_kernel(const float* __restrict__ qkv,
                               const float* __restrict__ scales,
                               _Float16* __restrict__ KT,
                               _Float16* __restrict__ VT) {
  const int total = NBATCH * NKV * SEQ * HEADD;
  const int stride = gridDim.x * blockDim.x;
  for (int i = blockIdx.x * blockDim.x + threadIdx.x; i < total; i += stride) {
    const int bk = i / (SEQ * HEADD);
    const int rem = i % (SEQ * HEADD);
    const int n = rem / HEADD, d = rem % HEADD;
    const int b = bk >> 2, kvh = bk & 3;
    const size_t row = (size_t)(b * SEQ + n) * NQKV;
    float kval = qkv[row + DIMC + kvh * HEADD + d];
    kval = fmaxf(kval, 0.f);
    const float kf = scales[NBATCH * NHEAD + bk] * kval * kval * kval;
    const size_t o = (size_t)bk * HEADD * SEQ + (size_t)d * SEQ + n;
    KT[o] = (_Float16)kf;
    VT[o] = (_Float16)qkv[row + DIMC + 384 + kvh * HEADD + d];
  }
}

// 3x3 depthwise conv on v over the 32x32 latent grid (SAME, zero pad).
__global__ void dwconv_kernel(const float* __restrict__ qkv,
                              const float* __restrict__ w,
                              const float* __restrict__ bias,
                              float* __restrict__ vdwc) {
  const int total = NBATCH * NKV * SEQ * HEADD;
  const int stride = gridDim.x * blockDim.x;
  for (int i = blockIdx.x * blockDim.x + threadIdx.x; i < total; i += stride) {
    const int bk = i / (SEQ * HEADD);
    const int rem = i % (SEQ * HEADD);
    const int n = rem / HEADD, e = rem % HEADD;
    const int b = bk >> 2, kvh = bk & 3;
    const int c = kvh * HEADD + e;
    const int y = n >> 5, x = n & 31;
    float s = bias[c];
#pragma unroll
    for (int dy = 0; dy < 3; ++dy) {
      const int yy = y + dy - 1;
      if (yy < 0 || yy > 31) continue;
#pragma unroll
      for (int dx = 0; dx < 3; ++dx) {
        const int xx = x + dx - 1;
        if (xx < 0 || xx > 31) continue;
        s += w[c * 9 + dy * 3 + dx] *
             qkv[((size_t)(b * SEQ) + yy * 32 + xx) * NQKV + DIMC + 384 + c];
      }
    }
    vdwc[i] = s;   // layout [bk][n][e]
  }
}

// ---------------------------------------------------------------------------
// Host launcher
// ---------------------------------------------------------------------------
extern "C" void kernel_launch(void* const* d_in, const int* in_sizes, int n_in,
                              void* d_out, int out_size, void* d_ws, size_t ws_size,
                              hipStream_t stream) {
  (void)in_sizes; (void)n_in; (void)out_size; (void)ws_size;
  const float* x      = (const float*)d_in[0];
  const float* wq_w   = (const float*)d_in[1];
  const float* wq_b   = (const float*)d_in[2];
  const float* wkv_w  = (const float*)d_in[3];
  const float* wkv_b  = (const float*)d_in[4];
  const float* dwc_w  = (const float*)d_in[5];
  const float* dwc_b  = (const float*)d_in[6];
  const float* proj_w = (const float*)d_in[7];
  const float* proj_b = (const float*)d_in[8];

  char* ws = (char*)d_ws;
  size_t off = 0;
  auto take = [&](size_t bytes) -> char* {
    char* p = ws + off;
    off += (bytes + 255) & ~(size_t)255;
    return p;
  };

  float*    qkv    = (float*)take((size_t)TOKENS * NQKV * 4);     // fp32 q|k|v
  _Float16* Xh     = (_Float16*)take((size_t)TOKENS * DIMC * 2);  // x f16, reused as Qh
  _Float16* WhT    = (_Float16*)take((size_t)NQKV * DIMC * 2);    // [out][in] f16
  _Float16* PhT    = (_Float16*)take((size_t)DIMC * DIMC * 2);    // proj [out][in] f16
  float*    biasC  = (float*)take(NQKV * 4);
  float*    scales = (float*)take(512 * 4);
  _Float16* KT     = (_Float16*)take((size_t)NBATCH * NKV * HEADD * SEQ * 2);
  _Float16* VT     = (_Float16*)take((size_t)NBATCH * NKV * HEADD * SEQ * 2);
  _Float16* kvT    = (_Float16*)take((size_t)NBATCH * NKV * HEADD * HEADD * 2);
  float*    vdwc   = (float*)take((size_t)NBATCH * NKV * SEQ * HEADD * 4);
  _Float16* Qh = Xh;              // Xh dead after the QKV GEMM
  _Float16* Oh = (_Float16*)qkv;  // qkv region dead after pack/conv kernels

  // 1. x -> f16
  f32_to_f16_vec<<<2048, 256, 0, stream>>>(x, Xh, TOKENS * DIMC / 4);
  // 2. weight packing (transposed so GEMM B staging is coalesced)
  transpose_to_f16<<<1024, 256, 0, stream>>>(wq_w, WhT, DIMC, DIMC);
  transpose_to_f16<<<1024, 256, 0, stream>>>(wkv_w, WhT + (size_t)DIMC * DIMC, DIMC, 768);
  transpose_to_f16<<<1024, 256, 0, stream>>>(proj_w, PhT, DIMC, DIMC);
  pack_bias_kernel<<<8, 256, 0, stream>>>(wq_b, wkv_b, biasC);
  // 3. fused q/kv projection: [32768,1152] x [1152,1920]
  gemm_f16_f32<<<dim3(NQKV / 128, TOKENS / 128), 256, 0, stream>>>(
      Xh, WhT, biasC, qkv, TOKENS, NQKV, DIMC);
  // 4. focus norms per (batch, head)
  reduce_scales_kernel<<<512, 256, 0, stream>>>(qkv, scales);
  // 5. focus map + f16 layout packing
  focus_q_kernel<<<4096, 256, 0, stream>>>(qkv, scales, Qh);
  pack_kv_kernel<<<4096, 256, 0, stream>>>(qkv, scales, KT, VT);
  // 6. kv = k^T v per (b, kv-head)
  gemm_kv_kernel<<<128, 192, 0, stream>>>(KT, VT, kvT);
  // 7. depthwise conv of v on 32x32 latent grid
  dwconv_kernel<<<4096, 256, 0, stream>>>(qkv, dwc_w, dwc_b, vdwc);
  // 8. attn = q @ kv, + v_dwc, repack f16 (overwrites qkv arena region)
  gemm_attn_kernel<<<dim3(SEQ / 128, NBATCH * NHEAD), 256, 0, stream>>>(
      Qh, kvT, vdwc, Oh);
  // 9. output projection: [32768,1152] x [1152,1152] + proj_b -> d_out fp32
  gemm_f16_f32<<<dim3(DIMC / 128, TOKENS / 128), 256, 0, stream>>>(
      Oh, PhT, proj_b, (float*)d_out, TOKENS, DIMC, DIMC);
}
